// Attention_20907900797348
// MI455X (gfx1250) — matrix operand order repaired and checked by reading, not compiled, not run on hardware
//
#include <hip/hip_runtime.h>
#include <hip/hip_bf16.h>
#include <stdint.h>

#define WINDOW 14
#define NTOK 196        // 14*14
#define NPAD2 224       // 7*32, padded K for the PV GEMM
#define NHEADS 12
#define DIM 768
#define HDIM 64
#define BATCH 64
#define SCALE 0.125f    // 64^-0.5

typedef __attribute__((ext_vector_type(16))) _Float16 v16h;
typedef __attribute__((ext_vector_type(8)))  _Float16 v8h;
typedef __attribute__((ext_vector_type(8)))  float    v8f;

union FragU { v8h h[2]; v16h f; };

// ---- CDNA5 async global->LDS copy (ASYNCcnt-tracked), SADDR form ----
// LDS byte address taken from the vdst VGPR; global = SGPR base + 32-bit unsigned voffset.
__device__ __forceinline__ void async_ld_b128(unsigned ldsaddr, unsigned voff, const void* sbase) {
    asm volatile("global_load_async_to_lds_b128 %0, %1, %2"
                 :: "v"(ldsaddr), "v"(voff), "s"((uint64_t)(uintptr_t)sbase)
                 : "memory");
}
__device__ __forceinline__ void wait_async0() { asm volatile("s_wait_asynccnt 0x0" ::: "memory"); }
__device__ __forceinline__ void wait_async4() { asm volatile("s_wait_asynccnt 0x4" ::: "memory"); }

__device__ __forceinline__ unsigned lds_addr(const void* p) {
    return (unsigned)(uintptr_t)p;   // low 32 bits of generic addr = LDS byte offset
}

// A fragment (16x32 f16): lanes 0-15 -> rows 0-15, K 0..7 & 16..23;
// lanes 16-31 -> rows 0-15, K 8..15 & 24..31. Two b128 loads.
__device__ __forceinline__ v16h load_frag_a(const _Float16* base, int stride, int lane, int k0) {
    int r  = lane & 15;
    int kk = k0 + ((lane >> 4) << 3);
    FragU u;
    u.h[0] = *reinterpret_cast<const v8h*>(base + r * stride + kk);
    u.h[1] = *reinterpret_cast<const v8h*>(base + r * stride + kk + 16);
    return u.f;
}

// B fragment (32x16 f16) from a [n][k] row-major buffer: lane -> col n = lane&15,
// K runs k0+(lane>>4)*8 .. +7 and +16 .. +23 (matches A). Two b128 loads.
__device__ __forceinline__ v16h load_frag_b(const _Float16* base, int stride, int lane, int k0) {
    int n  = lane & 15;
    int kk = k0 + ((lane >> 4) << 3);
    FragU u;
    u.h[0] = *reinterpret_cast<const v8h*>(base + n * stride + kk);
    u.h[1] = *reinterpret_cast<const v8h*>(base + n * stride + kk + 16);
    return u.f;
}

__device__ __forceinline__ v8f wmma_f16(v16h a, v16h b, v8f c) {
    return __builtin_amdgcn_wmma_f32_16x16x32_f16(false, a, false, b, (short)0, c, false, false);
}

// ---------------- kernel 0: weight conversion f32 -> f16 ----------------
__global__ void k_convert(const float* __restrict__ wqkv, const float* __restrict__ wproj,
                          _Float16* __restrict__ wqkv_h, _Float16* __restrict__ wproj_h) {
    const int n1 = 3 * DIM * DIM, n2 = DIM * DIM;
    for (int i = blockIdx.x * blockDim.x + threadIdx.x; i < n1 + n2; i += gridDim.x * blockDim.x) {
        if (i < n1) wqkv_h[i] = (_Float16)wqkv[i];
        else        wproj_h[i - n1] = (_Float16)wproj[i - n1];
    }
}

// ---------------- kernel 1: QKV GEMM (12544 x 2304 = X @ Wqkv^T) ----------------
// 128x128 tile per workgroup, 8 waves (4x2). W tile staged via async-to-LDS;
// X tile converted f32->f16 through VALU. Epilogue fuses bias + q scaling.
__global__ __launch_bounds__(256) void k_qkv(
        const float* __restrict__ x, const _Float16* __restrict__ wqkv_h,
        const float* __restrict__ q_bias, const float* __restrict__ v_bias,
        _Float16* __restrict__ qh, _Float16* __restrict__ kh, _Float16* __restrict__ vh) {
    __shared__ __align__(16) _Float16 Xs[128 * 32];
    __shared__ __align__(16) _Float16 Ws[128 * 32];
    const int t = threadIdx.x, lane = t & 31, w = t >> 5;
    const int wm = w >> 1, wn = w & 1;
    const int m0 = blockIdx.x * 128, n0 = blockIdx.y * 128;
    const int srow = t >> 1, sseg = t & 1;
    const unsigned wlds = lds_addr(&Ws[0]) + (unsigned)((srow * 32 + sseg * 16) * 2);

    const v8f vzero = {};
    v8f acc[2][4];
    #pragma unroll
    for (int it = 0; it < 2; ++it)
        #pragma unroll
        for (int jt = 0; jt < 4; ++jt) acc[it][jt] = vzero;

    for (int k0 = 0; k0 < DIM; k0 += 32) {
        // async-stage W tile: 128 rows x 32 halves, [n][k], 2 x b128 per thread
        {
            unsigned go = (unsigned)((((size_t)(n0 + srow) * DIM + k0 + sseg * 16)) * 2);
            async_ld_b128(wlds,      go,      wqkv_h);
            async_ld_b128(wlds + 16, go + 16, wqkv_h);
        }
        // stage X tile (f32 -> f16) through VALU cvt
        #pragma unroll
        for (int rpt = 0; rpt < 4; ++rpt) {
            int idx = rpt * 256 + t;              // 1024 float4 = 128 rows x 8
            int row = idx >> 3, c4 = idx & 7;
            float4 f = *reinterpret_cast<const float4*>(x + (size_t)(m0 + row) * DIM + k0 + c4 * 4);
            _Float16* dst = &Xs[row * 32 + c4 * 4];
            dst[0] = (_Float16)f.x; dst[1] = (_Float16)f.y;
            dst[2] = (_Float16)f.z; dst[3] = (_Float16)f.w;
        }
        wait_async0();
        __syncthreads();

        v16h a[2], b[4];
        #pragma unroll
        for (int it = 0; it < 2; ++it)
            a[it] = load_frag_a(Xs + (wm * 32 + it * 16) * 32, 32, lane, 0);
        #pragma unroll
        for (int jt = 0; jt < 4; ++jt)
            b[jt] = load_frag_b(Ws + (wn * 64 + jt * 16) * 32, 32, lane, 0);
        #pragma unroll
        for (int it = 0; it < 2; ++it)
            #pragma unroll
            for (int jt = 0; jt < 4; ++jt)
                acc[it][jt] = wmma_f16(a[it], b[jt], acc[it][jt]);
        __syncthreads();
    }

    // epilogue: C layout -> row = (lane>>4)*8 + i, col = lane&15 within each tile
    #pragma unroll
    for (int it = 0; it < 2; ++it) {
        int mbase0 = m0 + wm * 32 + it * 16 + ((lane >> 4) << 3);
        #pragma unroll
        for (int jt = 0; jt < 4; ++jt) {
            int c   = n0 + wn * 64 + jt * 16 + (lane & 15);
            int sec = c / DIM, cc = c % DIM;
            int hh  = cc >> 6, d = cc & 63;
            float qb = q_bias[cc];      // hoisted: one load per tile
            float vb = v_bias[cc];
            int bw = mbase0 / NTOK, tok = mbase0 % NTOK;
            #pragma unroll
            for (int i = 0; i < 8; ++i) {
                size_t off = ((size_t)(bw * NHEADS + hh) * NTOK + tok) * HDIM + d;
                float val = acc[it][jt][i];
                if (sec == 0)      qh[off] = (_Float16)((val + qb) * SCALE);
                else if (sec == 1) kh[off] = (_Float16)val;
                else               vh[off] = (_Float16)(val + vb);
                if (++tok == NTOK) { tok = 0; ++bw; }
            }
        }
    }
}

// ---------------- kernel 2: attention per (b,h), 13 waves, 16 queries/wave ----------------
__global__ __launch_bounds__(416) void k_attn(
        const _Float16* __restrict__ qh, const _Float16* __restrict__ kh,
        const _Float16* __restrict__ vh, const float* __restrict__ rpb_table,
        const int* __restrict__ rel_idx, _Float16* __restrict__ oh) {
    __shared__ __align__(16) _Float16 Ks[208 * HDIM];        // keys  [tok][d]
    __shared__ __align__(16) _Float16 Vt[HDIM * NPAD2];      // V^T   [d][tok]
    __shared__ __align__(16) _Float16 Ps[13 * 16 * NPAD2];   // P per-wave strips
    const int t = threadIdx.x, lane = t & 31, w = t >> 5;
    const int bh = blockIdx.x, b = bh / NHEADS, h = bh % NHEADS;
    const size_t base = (size_t)bh * NTOK * HDIM;

    // async-stage K (zero-pad rows 196..207 with DS stores)
    {
        int row = t >> 1, c0 = (t & 1) * 32;
        if (row < NTOK) {
            unsigned lk = lds_addr(&Ks[0]) + (unsigned)((row * HDIM + c0) * 2);
            unsigned gk = (unsigned)(((size_t)row * HDIM + c0) * 2);
            const void* kb = kh + base;
            async_ld_b128(lk,      gk,      kb);
            async_ld_b128(lk + 16, gk + 16, kb);
            async_ld_b128(lk + 32, gk + 32, kb);
            async_ld_b128(lk + 48, gk + 48, kb);
        } else {
            uint4 z = {0, 0, 0, 0};
            uint4* dst = reinterpret_cast<uint4*>(&Ks[row * HDIM + c0]);
            dst[0] = z; dst[1] = z; dst[2] = z; dst[3] = z;
        }
    }
    // stage V transposed into [d][tok] with zero-padded tokens 196..223
    for (int idx = t; idx < NTOK * HDIM; idx += 416) {
        int r = idx >> 6, d = idx & 63;
        Vt[d * NPAD2 + r] = vh[base + idx];
    }
    for (int idx = t; idx < HDIM * (NPAD2 - NTOK); idx += 416) {
        int d = idx / (NPAD2 - NTOK), r = NTOK + idx % (NPAD2 - NTOK);
        Vt[d * NPAD2 + r] = (_Float16)0.f;
    }
    wait_async0();
    __syncthreads();

    const int row0 = w * 16;
    // Q fragments straight from global (row-major [tok][d] == A layout runs)
    v16h aq[2];
    {
        int r = row0 + (lane & 15); if (r >= NTOK) r = NTOK - 1;
        const _Float16* qrow = qh + base + (size_t)r * HDIM;
        #pragma unroll
        for (int ks = 0; ks < 2; ++ks) {
            int kk = ks * 32 + ((lane >> 4) << 3);
            FragU u;
            u.h[0] = *reinterpret_cast<const v8h*>(qrow + kk);
            u.h[1] = *reinterpret_cast<const v8h*>(qrow + kk + 16);
            aq[ks] = u.f;
        }
    }

    // S = q k^T : 13 column tiles x (K=64 in two 32-steps)
    const v8f vzero = {};
    v8f s[13];
    #pragma unroll
    for (int jt = 0; jt < 13; ++jt) s[jt] = vzero;
    #pragma unroll
    for (int jt = 0; jt < 13; ++jt) {
        v16h b0 = load_frag_b(Ks + jt * 16 * HDIM, HDIM, lane, 0);
        v16h b1 = load_frag_b(Ks + jt * 16 * HDIM, HDIM, lane, 32);
        s[jt] = wmma_f16(aq[0], b0, s[jt]);
        s[jt] = wmma_f16(aq[1], b1, s[jt]);
    }

    // + relative position bias, mask pad columns
    const int colb = lane & 15;
    #pragma unroll
    for (int jt = 0; jt < 13; ++jt) {
        int mc = jt * 16 + colb;
        #pragma unroll
        for (int i = 0; i < 8; ++i) {
            int qr = row0 + ((lane >> 4) << 3) + i;
            float sv = s[jt][i];
            if (mc < NTOK && qr < NTOK)
                sv += rpb_table[rel_idx[qr * NTOK + mc] * NHEADS + h];
            if (mc >= NTOK) sv = -1e30f;
            s[jt][i] = sv;
        }
    }

    // register softmax: row r lives in VGPR i=r%8 across the 16 lanes of one half-wave
    float rinv[8];
    #pragma unroll
    for (int i = 0; i < 8; ++i) {
        float mx = -1e30f;
        #pragma unroll
        for (int jt = 0; jt < 13; ++jt) mx = fmaxf(mx, s[jt][i]);
        for (int off = 1; off < 16; off <<= 1) mx = fmaxf(mx, __shfl_xor(mx, off, 32));
        float sum = 0.f;
        #pragma unroll
        for (int jt = 0; jt < 13; ++jt) { float p = __expf(s[jt][i] - mx); s[jt][i] = p; sum += p; }
        for (int off = 1; off < 16; off <<= 1) sum += __shfl_xor(sum, off, 32);
        rinv[i] = 1.f / fmaxf(sum, 1e-20f);
    }

    // write un-normalized P (f16) to this wave's LDS strip; pad cols 208..223
    _Float16* Pw = Ps + w * 16 * NPAD2;
    #pragma unroll
    for (int jt = 0; jt < 13; ++jt)
        #pragma unroll
        for (int i = 0; i < 8; ++i)
            Pw[(((lane >> 4) << 3) + i) * NPAD2 + jt * 16 + colb] = (_Float16)s[jt][i];
    #pragma unroll
    for (int i = 0; i < 8; ++i)
        Pw[(((lane >> 4) << 3) + i) * NPAD2 + 208 + colb] = (_Float16)0.f;
    __syncthreads();

    // O = P V : K = 224 in 7 steps of 32, 4 output tiles (d = 64)
    v8f o[4];
    #pragma unroll
    for (int jn = 0; jn < 4; ++jn) o[jn] = vzero;
    for (int kt = 0; kt < 7; ++kt) {
        v16h a = load_frag_a(Pw, NPAD2, lane, kt * 32);
        #pragma unroll
        for (int jn = 0; jn < 4; ++jn) {
            v16h bb = load_frag_b(Vt + jn * 16 * NPAD2, NPAD2, lane, kt * 32);
            o[jn] = wmma_f16(a, bb, o[jn]);
        }
    }

    // normalize + store f16 [b][tok][h*64+d]
    #pragma unroll
    for (int jn = 0; jn < 4; ++jn) {
        int d = jn * 16 + colb;
        #pragma unroll
        for (int i = 0; i < 8; ++i) {
            int qr = row0 + ((lane >> 4) << 3) + i;
            if (qr < NTOK)
                oh[((size_t)(b * NTOK) + qr) * DIM + h * HDIM + d] = (_Float16)(o[jn][i] * rinv[i]);
        }
    }
}

// ---------------- kernel 3: projection GEMM (12544 x 768) + bias, f32 out ----------------
// Double-buffered async-to-LDS staging: next tile's loads in flight under WMMAs.
__global__ __launch_bounds__(256) void k_proj(
        const _Float16* __restrict__ ah, const _Float16* __restrict__ wp,
        const float* __restrict__ pb, float* __restrict__ out) {
    __shared__ __align__(16) _Float16 As[2][128 * 32];
    __shared__ __align__(16) _Float16 Bs[2][128 * 32];
    const int t = threadIdx.x, lane = t & 31, w = t >> 5;
    const int wm = w >> 1, wn = w & 1;
    const int m0 = blockIdx.x * 128, n0 = blockIdx.y * 128;
    const int srow = t >> 1, sseg = t & 1;
    const unsigned albase = lds_addr(&As[0][0]) + (unsigned)((srow * 32 + sseg * 16) * 2);
    const unsigned blbase = lds_addr(&Bs[0][0]) + (unsigned)((srow * 32 + sseg * 16) * 2);
    const unsigned bufstep = 128 * 32 * 2;   // bytes per buffer
    const unsigned ga0 = (unsigned)(((size_t)(m0 + srow) * DIM + sseg * 16) * 2);
    const unsigned gb0 = (unsigned)(((size_t)(n0 + srow) * DIM + sseg * 16) * 2);

    const v8f vzero = {};
    v8f acc[2][4];
    #pragma unroll
    for (int it = 0; it < 2; ++it)
        #pragma unroll
        for (int jt = 0; jt < 4; ++jt) acc[it][jt] = vzero;

    // prologue: stage tile 0 into buffer 0
    {
        async_ld_b128(albase,      ga0,      ah);
        async_ld_b128(albase + 16, ga0 + 16, ah);
        async_ld_b128(blbase,      gb0,      wp);
        async_ld_b128(blbase + 16, gb0 + 16, wp);
    }

    for (int kt = 0; kt < 24; ++kt) {
        int cur = kt & 1;
        if (kt < 23) {   // stage tile kt+1 into the other buffer
            unsigned lo = (unsigned)((cur ^ 1)) * bufstep;
            unsigned go = (unsigned)((kt + 1) * 32 * 2);
            async_ld_b128(albase + lo,      ga0 + go,      ah);
            async_ld_b128(albase + lo + 16, ga0 + go + 16, ah);
            async_ld_b128(blbase + lo,      gb0 + go,      wp);
            async_ld_b128(blbase + lo + 16, gb0 + go + 16, wp);
            wait_async4();   // current tile complete; next tile (4 loads) in flight
        } else {
            wait_async0();
        }
        __syncthreads();

        v16h a[2], b[4];
        #pragma unroll
        for (int it = 0; it < 2; ++it)
            a[it] = load_frag_a(&As[cur][(wm * 32 + it * 16) * 32], 32, lane, 0);
        #pragma unroll
        for (int jt = 0; jt < 4; ++jt)
            b[jt] = load_frag_b(&Bs[cur][(wn * 64 + jt * 16) * 32], 32, lane, 0);
        #pragma unroll
        for (int it = 0; it < 2; ++it)
            #pragma unroll
            for (int jt = 0; jt < 4; ++jt)
                acc[it][jt] = wmma_f16(a[it], b[jt], acc[it][jt]);
        __syncthreads();   // readers done before buffer is overwritten next iter
    }

    #pragma unroll
    for (int it = 0; it < 2; ++it) {
        #pragma unroll
        for (int jt = 0; jt < 4; ++jt) {
            int c = n0 + wn * 64 + jt * 16 + (lane & 15);
            float bias = pb[c];
            #pragma unroll
            for (int i = 0; i < 8; ++i) {
                int m = m0 + wm * 32 + it * 16 + ((lane >> 4) << 3) + i;
                out[(size_t)m * DIM + c] = acc[it][jt][i] + bias;
            }
        }
    }
}

extern "C" void kernel_launch(void* const* d_in, const int* in_sizes, int n_in,
                              void* d_out, int out_size, void* d_ws, size_t ws_size,
                              hipStream_t stream) {
    (void)in_sizes; (void)n_in; (void)out_size; (void)ws_size;
    const float* x    = (const float*)d_in[0];
    const float* wqkv = (const float*)d_in[1];
    const float* qb   = (const float*)d_in[2];
    const float* vb   = (const float*)d_in[3];
    const float* rpb  = (const float*)d_in[4];
    const int*   ridx = (const int*)d_in[5];
    const float* wp   = (const float*)d_in[6];
    const float* pb   = (const float*)d_in[7];
    float* out = (float*)d_out;

    _Float16* ws = (_Float16*)d_ws;
    size_t o = 0;
    _Float16* wqkv_h = ws + o; o += (size_t)3 * DIM * DIM;
    _Float16* wp_h   = ws + o; o += (size_t)DIM * DIM;
    const size_t per = (size_t)BATCH * NHEADS * NTOK * HDIM;  // 9,633,792 halves
    _Float16* qh = ws + o; o += per;
    _Float16* kh = ws + o; o += per;
    _Float16* vh = ws + o; o += per;
    _Float16* ah = ws + o; o += per;

    k_convert<<<2048, 256, 0, stream>>>(wqkv, wp, wqkv_h, wp_h);
    k_qkv<<<dim3(98, 18), 256, 0, stream>>>(x, wqkv_h, qb, vb, qh, kh, vh);
    k_attn<<<BATCH * NHEADS, 416, 0, stream>>>(qh, kh, vh, rpb, ridx, ah);
    k_proj<<<dim3(98, 6), 256, 0, stream>>>(ah, wp_h, pb, out);
}
